// EfficientAdditiveAttention_89455578841351
// MI455X (gfx1250) — matrix operand order repaired and checked by reading, not compiled, not run on hardware
//
#include <hip/hip_runtime.h>
#include <hip/hip_bf16.h>

// ---------------------------------------------------------------------------
// EfficientAdditiveAttention on MI455X (gfx1250)
// B=8, N=8192, IN=512, TOTAL=512, H=8, D=64
// bf16 WMMA (16x16x32) for all GEMMs, f32 accumulate / norm / softmax.
// ---------------------------------------------------------------------------

#define B_   8
#define N_   8192
#define CIN  512
#define TOT  512
#define NH   8
#define HD   64
#define BN   (B_ * N_)

typedef __attribute__((ext_vector_type(16))) __bf16 v16bf;
typedef __attribute__((ext_vector_type(8)))  float  v8f;

union FragBF {
    v16bf v;
    uint4 q[2];
};

// A-matrix fragment (16x32 bf16), row-major source with leading dim ld.
// lane L: M = L&15, holds K = k0 + 8*(L>>4) + {0..7}  (VGPR0-3)
//                    and  K = k0 + 16 + 8*(L>>4) + {0..7} (VGPR4-7)
__device__ __forceinline__ v16bf load_a_frag(const __bf16* base, int row, int ld,
                                             int k0, int hi) {
    FragBF f;
    const __bf16* p = base + (size_t)row * ld + k0 + 8 * hi;
    f.q[0] = *(const uint4*)(p);
    f.q[1] = *(const uint4*)(p + 16);
    return f.v;
}

// B-matrix fragment (32x16 bf16) sourced from W row-major: B[k][n] = W[n][k].
// lane L: N = L&15, holds K = k0 + 8*(L>>4) + {0..7} and K = k0 + 16 + 8*(L>>4) + {0..7}
__device__ __forceinline__ v16bf load_b_frag(const __bf16* base, int wrow, int ld,
                                             int k0, int hi) {
    FragBF f;
    const __bf16* p = base + (size_t)wrow * ld + k0 + 8 * hi;
    f.q[0] = *(const uint4*)(p);
    f.q[1] = *(const uint4*)(p + 16);
    return f.v;
}

__device__ __forceinline__ v8f wmma_bf16(v16bf a, v16bf b, v8f c) {
    return __builtin_amdgcn_wmma_f32_16x16x32_bf16(
        /*neg_a=*/false, a, /*neg_b=*/false, b,
        /*c_mod=*/(short)0, c, /*reuse_a=*/false, /*reuse_b=*/false);
}

// ---------------------------------------------------------------------------
// Kernel 0: f32 -> bf16 conversion (x and weights)
// ---------------------------------------------------------------------------
__global__ void k0_convert(const float* __restrict__ src, __bf16* __restrict__ dst,
                           int n) {
    int i = blockIdx.x * 256 + threadIdx.x;
    if (i < n) dst[i] = (__bf16)src[i];
}

// ---------------------------------------------------------------------------
// Kernel 1: q/k head projections + L2 norm + gating logits.
// grid = (BN/64, NH), block = 256 (8 waves).
// waves 0-3: q rows strip; waves 4-7: k rows strip. x fragments hit L2.
// ---------------------------------------------------------------------------
__global__ __launch_bounds__(256) void k1_qk(
    const __bf16* __restrict__ xbf, const __bf16* __restrict__ Wqb,
    const __bf16* __restrict__ Wkb, const float* __restrict__ bq,
    const float* __restrict__ bk, const float* __restrict__ w_g,
    float* __restrict__ qn, float* __restrict__ kn,
    float* __restrict__ logits) {
    const int tile = blockIdx.x;   // 64-token tile
    const int head = blockIdx.y;
    const int lane = threadIdx.x & 31;
    const int wave = threadIdx.x >> 5;
    const bool isQ = (wave < 4);
    const int rs = wave & 3;
    const int mbase = tile * 64 + rs * 16;
    const int hi = lane >> 4;
    const int l15 = lane & 15;

    const __bf16* W   = isQ ? Wqb : Wkb;
    const float* bias = isQ ? bq : bk;
    float* outp       = isQ ? qn : kn;

    v8f acc[4] = {};
    const int arow = mbase + l15;
#pragma unroll 4
    for (int k0 = 0; k0 < CIN; k0 += 32) {
        v16bf a = load_a_frag(xbf, arow, CIN, k0, hi);
#pragma unroll
        for (int nt = 0; nt < 4; ++nt) {
            v16bf b = load_b_frag(W, head * HD + nt * 16 + l15, CIN, k0, hi);
            acc[nt] = wmma_bf16(a, b, acc[nt]);
        }
    }
    // bias (zeros in this config, kept for generality)
#pragma unroll
    for (int nt = 0; nt < 4; ++nt) {
        float bv = bias[head * HD + nt * 16 + l15];
#pragma unroll
        for (int j = 0; j < 8; ++j) acc[nt][j] += bv;
    }
    // L2 normalize each row (64 cols spread over 16 lanes x 4 tiles)
#pragma unroll
    for (int j = 0; j < 8; ++j) {
        float s = acc[0][j] * acc[0][j] + acc[1][j] * acc[1][j] +
                  acc[2][j] * acc[2][j] + acc[3][j] * acc[3][j];
        s += __shfl_xor(s, 1, 32);
        s += __shfl_xor(s, 2, 32);
        s += __shfl_xor(s, 4, 32);
        s += __shfl_xor(s, 8, 32);
        float inv = 1.0f / fmaxf(sqrtf(s), 1e-12f);
#pragma unroll
        for (int nt = 0; nt < 4; ++nt) acc[nt][j] *= inv;
    }
    // store normalized head + (q only) gating logit
#pragma unroll
    for (int j = 0; j < 8; ++j) {
        const int row = mbase + j + 8 * hi;  // flat b*N + n
        const size_t base = (size_t)row * TOT + head * HD;
#pragma unroll
        for (int nt = 0; nt < 4; ++nt) outp[base + nt * 16 + l15] = acc[nt][j];
        if (isQ) {
            float lg = acc[0][j] * w_g[head * HD + 0 * 16 + l15] +
                       acc[1][j] * w_g[head * HD + 1 * 16 + l15] +
                       acc[2][j] * w_g[head * HD + 2 * 16 + l15] +
                       acc[3][j] * w_g[head * HD + 3 * 16 + l15];
            lg += __shfl_xor(lg, 1, 32);
            lg += __shfl_xor(lg, 2, 32);
            lg += __shfl_xor(lg, 4, 32);
            lg += __shfl_xor(lg, 8, 32);
            if (l15 == 0) {
                int bb = row >> 13;       // N = 8192
                int nn = row & (N_ - 1);
                logits[((size_t)bb * NH + head) * N_ + nn] = lg * 0.125f; // 64^-0.5
            }
        }
    }
}

// ---------------------------------------------------------------------------
// Kernel 2: softmax over N per (b,h) + pooled G[b,h,d] = sum_n A * qn.
// grid = B*NH, block = 256.
// ---------------------------------------------------------------------------
__global__ __launch_bounds__(256) void k2_softmax_g(
    const float* __restrict__ logits, const float* __restrict__ qn,
    float* __restrict__ G) {
    const int bh = blockIdx.x;
    const int b = bh >> 3, h = bh & 7;
    const float* lg = logits + (size_t)bh * N_;
    __shared__ float red[256];
    __shared__ float gred[4 * HD];
    const int t = threadIdx.x;

    float m = -3.0e38f;
    for (int n = t; n < N_; n += 256) m = fmaxf(m, lg[n]);
    red[t] = m;
    __syncthreads();
    for (int s = 128; s > 0; s >>= 1) {
        if (t < s) red[t] = fmaxf(red[t], red[t + s]);
        __syncthreads();
    }
    m = red[0];
    __syncthreads();

    float sum = 0.0f;
    for (int n = t; n < N_; n += 256) sum += __expf(lg[n] - m);
    red[t] = sum;
    __syncthreads();
    for (int s = 128; s > 0; s >>= 1) {
        if (t < s) red[t] += red[t + s];
        __syncthreads();
    }
    const float invs = 1.0f / red[0];

    const int d = t & (HD - 1);
    const int grp = t >> 6;  // 0..3
    float g = 0.0f;
    for (int n = grp; n < N_; n += 4) {
        float p = __expf(lg[n] - m);
        g += p * qn[((size_t)b * N_ + n) * TOT + h * HD + d];
    }
    gred[grp * HD + d] = g;
    __syncthreads();
    if (t < HD) {
        float v = (gred[t] + gred[HD + t] + gred[2 * HD + t] + gred[3 * HD + t]) * invs;
        G[(size_t)bh * HD + t] = v;
    }
}

// ---------------------------------------------------------------------------
// Kernel 3: out = ((G .* kn) @ Wp^T + bp + qn) @ Wf^T + bf
// grid = BN/64, block = 256 (8 waves). 64KB LDS tile reused for t then u.
// ---------------------------------------------------------------------------
__global__ __launch_bounds__(256) void k3_out(
    const float* __restrict__ kn, const float* __restrict__ qn,
    const float* __restrict__ G, const __bf16* __restrict__ Wpb,
    const __bf16* __restrict__ Wfb, const float* __restrict__ bp,
    const float* __restrict__ bfv, float* __restrict__ out) {
    __shared__ __bf16 ldsT[64 * TOT];  // 64 KB: t tile, then u tile
    const int tile = blockIdx.x;
    const int rowBase = tile * 64;
    const int b = rowBase >> 13;

    // stage t = G(b, :) .* kn(rows, :) as bf16
    for (int idx = threadIdx.x; idx < 64 * TOT; idx += 256) {
        int r = idx >> 9, c = idx & (TOT - 1);
        float g = G[(size_t)b * TOT + c];
        ldsT[idx] = (__bf16)(g * kn[((size_t)rowBase + r) * TOT + c]);
    }
    __syncthreads();

    const int lane = threadIdx.x & 31;
    const int wave = threadIdx.x >> 5;
    const int hi = lane >> 4, l15 = lane & 15;
    const int rs = wave & 3;           // row strip 0..3
    const int mb = rs * 16;            // local row base
    const int chalf = wave >> 2;       // col half: 0 -> cols 0..255, 1 -> 256..511

    // GEMM1: u(64x512) = t @ Wp^T ; each wave: 16 rows x 256 cols (16 tiles)
    v8f acc[16] = {};
#pragma unroll 2
    for (int k0 = 0; k0 < TOT; k0 += 32) {
        v16bf a = load_a_frag(ldsT, mb + l15, TOT, k0, hi);
#pragma unroll
        for (int ct = 0; ct < 16; ++ct) {
            v16bf bfrag = load_b_frag(Wpb, chalf * 256 + ct * 16 + l15, TOT, k0, hi);
            acc[ct] = wmma_bf16(a, bfrag, acc[ct]);
        }
    }
    __syncthreads();  // all reads of t tile complete

    // u = acc + bp + qn residual; restage as bf16
#pragma unroll
    for (int ct = 0; ct < 16; ++ct) {
        const int col = chalf * 256 + ct * 16 + l15;
        const float bpc = bp[col];
#pragma unroll
        for (int j = 0; j < 8; ++j) {
            const int lrow = mb + j + 8 * hi;
            float u = acc[ct][j] + bpc + qn[((size_t)rowBase + lrow) * TOT + col];
            ldsT[lrow * TOT + col] = (__bf16)u;
        }
    }
    __syncthreads();

    // GEMM2: out(64x64) = u @ Wf^T ; each wave: 16 rows x 32 cols (2 tiles)
    v8f oc[2] = {};
    const int ctb = chalf * 2;
#pragma unroll 4
    for (int k0 = 0; k0 < TOT; k0 += 32) {
        v16bf a = load_a_frag(ldsT, mb + l15, TOT, k0, hi);
#pragma unroll
        for (int tt = 0; tt < 2; ++tt) {
            v16bf bb = load_b_frag(Wfb, (ctb + tt) * 16 + l15, TOT, k0, hi);
            oc[tt] = wmma_bf16(a, bb, oc[tt]);
        }
    }
#pragma unroll
    for (int tt = 0; tt < 2; ++tt) {
        const int col = (ctb + tt) * 16 + l15;
        const float bc = bfv[col];
#pragma unroll
        for (int j = 0; j < 8; ++j) {
            const int lrow = mb + j + 8 * hi;
            out[((size_t)rowBase + lrow) * HD + col] = oc[tt][j] + bc;
        }
    }
}

// ---------------------------------------------------------------------------
// Host launcher
// ---------------------------------------------------------------------------
extern "C" void kernel_launch(void* const* d_in, const int* in_sizes, int n_in,
                              void* d_out, int out_size, void* d_ws, size_t ws_size,
                              hipStream_t stream) {
    const float* x   = (const float*)d_in[0];
    const float* Wq  = (const float*)d_in[1];
    const float* bq  = (const float*)d_in[2];
    const float* Wk  = (const float*)d_in[3];
    const float* bk  = (const float*)d_in[4];
    const float* w_g = (const float*)d_in[5];
    const float* Wp  = (const float*)d_in[6];
    const float* bp  = (const float*)d_in[7];
    const float* Wf  = (const float*)d_in[8];
    const float* bfv = (const float*)d_in[9];
    float* out = (float*)d_out;

    // workspace carve-up (all offsets 256B aligned)
    char* ws = (char*)d_ws;
    size_t off = 0;
    __bf16* xbf = (__bf16*)(ws + off); off += (size_t)BN * CIN * 2;          // 67,108,864
    __bf16* Wqb = (__bf16*)(ws + off); off += (size_t)TOT * CIN * 2;         //    524,288
    __bf16* Wkb = (__bf16*)(ws + off); off += (size_t)TOT * CIN * 2;
    __bf16* Wpb = (__bf16*)(ws + off); off += (size_t)TOT * TOT * 2;
    __bf16* Wfb = (__bf16*)(ws + off); off += (size_t)HD * TOT * 2;          //     65,536
    float* qn     = (float*)(ws + off); off += (size_t)BN * TOT * 4;         // 134,217,728
    float* kn     = (float*)(ws + off); off += (size_t)BN * TOT * 4;
    float* logits = (float*)(ws + off); off += (size_t)B_ * NH * N_ * 4;     //  2,097,152
    float* G      = (float*)(ws + off); off += (size_t)B_ * NH * HD * 4;     //     16,384
    (void)ws_size; (void)n_in; (void)in_sizes; (void)out_size;

    // 0) bf16 conversions
    {
        int n = BN * CIN;
        k0_convert<<<(n + 255) / 256, 256, 0, stream>>>(x, xbf, n);
        n = TOT * CIN;
        k0_convert<<<(n + 255) / 256, 256, 0, stream>>>(Wq, Wqb, n);
        k0_convert<<<(n + 255) / 256, 256, 0, stream>>>(Wk, Wkb, n);
        n = TOT * TOT;
        k0_convert<<<(n + 255) / 256, 256, 0, stream>>>(Wp, Wpb, n);
        n = HD * TOT;
        k0_convert<<<(n + 255) / 256, 256, 0, stream>>>(Wf, Wfb, n);
    }
    // 1) q/k projections + L2 norm + logits
    k1_qk<<<dim3(BN / 64, NH), 256, 0, stream>>>(xbf, Wqb, Wkb, bq, bk, w_g,
                                                 qn, kn, logits);
    // 2) softmax over tokens + pooled G
    k2_softmax_g<<<B_ * NH, 256, 0, stream>>>(logits, qn, G);
    // 3) gate * kn -> Wp + residual -> Wf
    k3_out<<<BN / 64, 256, 0, stream>>>(kn, qn, G, Wpb, Wfb, bp, bfv, out);
}
